// LocalNodeAttentionMultiHead_32959579030123
// MI455X (gfx1250) — compile-verified, matrix operand order checked by reading, not run on hardware
//
#include <hip/hip_runtime.h>
#include <hip/hip_bf16.h>

// ---------------------------------------------------------------------------
// LocalNodeAttentionMultiHead for MI455X (gfx1250, wave32, WMMA)
//
// Shapes: B=8, C=1024, T=32, H=W=7 (HW=49), nH=8, d=128, K_WIN=9, S=24
// Collapsed algorithm:
//   Kp = WkFull @ X   (1024x1024 x 12544)   [WMMA f16->f32, 2x2 reg blocking]
//   Qp = WqFull @ X   (shares B operand with Kp)
//   Vp = Wv[n] @ Kp   (8 x 128x128 x 1568)  [WMMA f16->f32]
//   attention: softmax over 9 temporal taps (wave per query, VALU+DS shfl)
//   Z  = WoFull @ Y + bo.sum(0) + 8*x_item  (1024x1024 x 9408) [WMMA, 2x4]
// ---------------------------------------------------------------------------

typedef _Float16 h16;
typedef __attribute__((ext_vector_type(16))) _Float16 v16h;
typedef __attribute__((ext_vector_type(8)))  float    v8f;
typedef __attribute__((ext_vector_type(4)))  float    v4f;

#define WMMA_F16(a, b, c) \
  __builtin_amdgcn_wmma_f32_16x16x32_f16(false, (a), false, (b), (short)0, (c), false, false)

namespace cfg {
enum {
  B = 8, C = 1024, T = 32, HW = 49, NH = 8, D = 128, KW = 9, S = 24,
  P1 = B * T * HW,          // 12544 projection pixels
  NT1 = P1 / 16,            // 784
  P2 = B * S * HW,          // 9408 output pixels
  NT2 = P2 / 16,            // 588
  QT = (T * HW) / 16,       // 98 column-tiles per (b,n) for V-GEMM
  MT = C / 16,              // 64 row tiles
  KB = C / 32               // 32 k-steps of 32
};
}

// ---- workspace layout (bytes, all 256-aligned) ----------------------------
struct WsOff {
  static const size_t ApkK  = 0;                                   // 64*32*32*16 f16 = 2 MiB
  static const size_t ApkQ  = ApkK  + 2097152;
  static const size_t ApkV  = ApkQ  + 2097152;                     // 8*8*4*32*16 f16 = 256 KiB
  static const size_t ApkO  = ApkV  + 262144;
  static const size_t BoSum = ApkO  + 2097152;                     // 1024 f32
  static const size_t Xpk   = BoSum + 4096;                        // 784*32*32*16 f16
  static const size_t Kp    = Xpk   + 25690112;                    // 12544*1024 f32
  static const size_t Qp    = Kp    + 51380224;
  static const size_t Vp    = Qp    + 51380224;
  static const size_t Ypk   = Vp    + 51380224;                    // 588*32*32*16 f16
  static const size_t Total = Ypk   + 19267584;                    // ~206 MB
};

// Frag packing convention (matches ISA 16-bit A/B layouts, wave32):
//   frag index f = (tile*KB + kb)*32 + lane ; 16 halves per frag
//   lane: m|n = lane&15, hi = lane>>4
//   element i: k = kb*32 + hi*8 + i + (i>=8 ? 8 : 0)
__device__ __forceinline__ int frag_k(int kb, int hi, int i) {
  return kb * 32 + hi * 8 + i + ((i >= 8) ? 8 : 0);
}

// ---------------------------------------------------------------------------
// Packing kernels (run once per launch; bandwidth-trivial)
// ---------------------------------------------------------------------------
__global__ void pack_wkq_kernel(const float* __restrict__ Wq,
                                const float* __restrict__ Wk,
                                h16* __restrict__ ApkQ, h16* __restrict__ ApkK) {
  int tid = blockIdx.x * blockDim.x + threadIdx.x;
  if (tid >= 64 * 32 * 32 * 16) return;
  int i  = tid & 15;
  int l  = (tid >> 4) & 31;
  int kb = (tid >> 9) & 31;
  int mt = tid >> 14;
  int m = l & 15, hi = l >> 4;
  int k = frag_k(kb, hi, i);
  int src = (mt * 16 + m) * cfg::C + k;
  ApkQ[tid] = (h16)Wq[src];
  ApkK[tid] = (h16)Wk[src];
}

__global__ void pack_wv_kernel(const float* __restrict__ Wv, h16* __restrict__ ApkV) {
  int tid = blockIdx.x * blockDim.x + threadIdx.x;
  if (tid >= 8 * 8 * 4 * 32 * 16) return;
  int i  = tid & 15;
  int l  = (tid >> 4) & 31;
  int kb = (tid >> 9) & 3;
  int mt = (tid >> 11) & 7;
  int n  = tid >> 14;
  int m = l & 15, hi = l >> 4;
  int k = frag_k(kb, hi, i);
  int src = n * cfg::D * cfg::D + (mt * 16 + m) * cfg::D + k;
  ApkV[tid] = (h16)Wv[src];
}

__global__ void pack_wo_kernel(const float* __restrict__ Wo, h16* __restrict__ ApkO) {
  int tid = blockIdx.x * blockDim.x + threadIdx.x;
  if (tid >= 64 * 32 * 32 * 16) return;
  int i  = tid & 15;
  int l  = (tid >> 4) & 31;
  int kb = (tid >> 9) & 31;
  int mt = tid >> 14;
  int m = l & 15, hi = l >> 4;
  int k = frag_k(kb, hi, i);             // k = n*128 + dd
  int c = mt * 16 + m;
  int src = ((k >> 7) * cfg::C + c) * cfg::D + (k & 127);
  ApkO[tid] = (h16)Wo[src];
}

__global__ void bo_sum_kernel(const float* __restrict__ bo, float* __restrict__ boSum) {
  int c = blockIdx.x * blockDim.x + threadIdx.x;
  if (c >= cfg::C) return;
  float s = 0.f;
#pragma unroll
  for (int n = 0; n < cfg::NH; ++n) s += bo[n * cfg::C + c];
  boSum[c] = s;
}

__global__ void pack_x_kernel(const float* __restrict__ x, h16* __restrict__ Xpk) {
  int tid = blockIdx.x * blockDim.x + threadIdx.x;
  if (tid >= cfg::NT1 * 32 * 32 * 16) return;
  int i  = tid & 15;
  int l  = (tid >> 4) & 31;
  int kb = (tid >> 9) & 31;
  int nt = tid >> 14;
  int ncol = l & 15, hi = l >> 4;
  int k = frag_k(kb, hi, i);             // channel c
  int p = nt * 16 + ncol;                // pixel (b,t,hw)
  int b  = p / (cfg::T * cfg::HW);
  int q  = p - b * (cfg::T * cfg::HW);
  int t  = q / cfg::HW;
  int hw = q - t * cfg::HW;
  size_t src = (((size_t)b * cfg::C + k) * cfg::T + t) * cfg::HW + hw;
  Xpk[tid] = (h16)x[src];
}

// ---------------------------------------------------------------------------
// Epilogue helpers
// ---------------------------------------------------------------------------
__device__ __forceinline__ void store_proj_tile(float* __restrict__ dst,
                                                const float* __restrict__ bias,
                                                const v8f& acc, int mt, int nt, int lane) {
  int hi  = lane >> 4;
  int m0  = mt * 16 + hi * 8;            // global row (= n*128+dd)
  int col = nt * 16 + (lane & 15);       // pixel (b,t,hw)
  int b   = col / (cfg::T * cfg::HW);
  int q   = col - b * (cfg::T * cfg::HW);
  int t   = q / cfg::HW;
  int hw  = q - t * cfg::HW;
  int nh  = mt >> 3;
  int dd0 = (mt & 7) * 16 + hi * 8;
  size_t base = (((size_t)(b * cfg::NH + nh) * cfg::T + t) * cfg::HW + hw) * cfg::D + dd0;
#pragma unroll
  for (int j = 0; j < 8; ++j) dst[base + j] = acc[j] + bias[m0 + j];
}

__device__ __forceinline__ void store_out_tile(float* __restrict__ out,
                                               const float* __restrict__ x,
                                               const float* __restrict__ boSum,
                                               const v8f& acc, int mt, int nt2, int lane) {
  int hi = lane >> 4;
  int p2 = nt2 * 16 + (lane & 15);
  int b  = p2 / (cfg::S * cfg::HW);
  int r  = p2 - b * (cfg::S * cfg::HW);
  int s  = r / cfg::HW;
  int hw = r - s * cfg::HW;
  int c0 = mt * 16 + hi * 8;
#pragma unroll
  for (int j = 0; j < 8; ++j) {
    int c = c0 + j;
    float resid = x[(((size_t)b * cfg::C + c) * cfg::T + (s + 4)) * cfg::HW + hw];
    out[(((size_t)b * cfg::C + c) * cfg::S + s) * cfg::HW + hw] =
        acc[j] + boSum[c] + 8.0f * resid;
  }
}

// ---------------------------------------------------------------------------
// GEMM 1: Kp = WkFull@X + bk, Qp = WqFull@X + bq
// 2 M-tiles x 2 N-tiles per wave, K and Q share B-frags:
//   per k-step: 6 frag loads -> 8 WMMAs.
// grid (392, 4), block 256 (8 waves)
// ---------------------------------------------------------------------------
__global__ __launch_bounds__(256) void gemm_kq_kernel(
    const h16* __restrict__ ApkK, const h16* __restrict__ ApkQ,
    const h16* __restrict__ Xpk,
    const float* __restrict__ bk, const float* __restrict__ bq,
    float* __restrict__ Kp, float* __restrict__ Qp) {
  int lane = threadIdx.x & 31;
  int w    = threadIdx.x >> 5;
  int mtp  = blockIdx.y * 8 + w;         // 0..31
  int mt0  = mtp * 2, mt1 = mt0 + 1;
  int nt0  = blockIdx.x * 2, nt1 = nt0 + 1;

  const v16h* Ak0 = ((const v16h*)ApkK) + (size_t)mt0 * (cfg::KB * 32) + lane;
  const v16h* Ak1 = ((const v16h*)ApkK) + (size_t)mt1 * (cfg::KB * 32) + lane;
  const v16h* Aq0 = ((const v16h*)ApkQ) + (size_t)mt0 * (cfg::KB * 32) + lane;
  const v16h* Aq1 = ((const v16h*)ApkQ) + (size_t)mt1 * (cfg::KB * 32) + lane;
  const v16h* Bx0 = ((const v16h*)Xpk)  + (size_t)nt0 * (cfg::KB * 32) + lane;
  const v16h* Bx1 = ((const v16h*)Xpk)  + (size_t)nt1 * (cfg::KB * 32) + lane;

  v8f aK00 = {0.f,0.f,0.f,0.f,0.f,0.f,0.f,0.f}, aK01 = aK00, aK10 = aK00, aK11 = aK00;
  v8f aQ00 = aK00, aQ01 = aK00, aQ10 = aK00, aQ11 = aK00;

#pragma unroll 2
  for (int kb = 0; kb < cfg::KB; ++kb) {
    v16h b0  = Bx0[kb * 32];
    v16h b1  = Bx1[kb * 32];
    v16h ak0 = Ak0[kb * 32];
    v16h ak1 = Ak1[kb * 32];
    v16h aq0 = Aq0[kb * 32];
    v16h aq1 = Aq1[kb * 32];
    aK00 = WMMA_F16(ak0, b0, aK00);
    aK01 = WMMA_F16(ak0, b1, aK01);
    aK10 = WMMA_F16(ak1, b0, aK10);
    aK11 = WMMA_F16(ak1, b1, aK11);
    aQ00 = WMMA_F16(aq0, b0, aQ00);
    aQ01 = WMMA_F16(aq0, b1, aQ01);
    aQ10 = WMMA_F16(aq1, b0, aQ10);
    aQ11 = WMMA_F16(aq1, b1, aQ11);
  }

  store_proj_tile(Kp, bk, aK00, mt0, nt0, lane);
  store_proj_tile(Kp, bk, aK01, mt0, nt1, lane);
  store_proj_tile(Kp, bk, aK10, mt1, nt0, lane);
  store_proj_tile(Kp, bk, aK11, mt1, nt1, lane);
  store_proj_tile(Qp, bq, aQ00, mt0, nt0, lane);
  store_proj_tile(Qp, bq, aQ01, mt0, nt1, lane);
  store_proj_tile(Qp, bq, aQ10, mt1, nt0, lane);
  store_proj_tile(Qp, bq, aQ11, mt1, nt1, lane);
}

// ---------------------------------------------------------------------------
// GEMM 2: Vp[n] = Wv[n] @ Kp[n] + bv   (B-frag built from f32 Kp on the fly)
// grid (98, 64), block 256 (wave = mtile 0..7)
// ---------------------------------------------------------------------------
__global__ __launch_bounds__(256) void gemm_v_kernel(
    const h16* __restrict__ ApkV, const float* __restrict__ Kp,
    const float* __restrict__ bv, float* __restrict__ Vp) {
  int lane = threadIdx.x & 31;
  int mt   = threadIdx.x >> 5;           // 0..7 (rows of 128)
  int qt   = blockIdx.x;                 // 0..97
  int bn   = blockIdx.y;                 // 0..63
  int n    = bn & 7;

  const v16h* Av = ((const v16h*)ApkV) + ((size_t)(n * 8 + mt) * 4) * 32 + lane;
  int hi   = lane >> 4;
  int colq = qt * 16 + (lane & 15);      // t*49+hw within (b,n)
  const float* kcol = Kp + ((size_t)bn * (cfg::T * cfg::HW) + colq) * cfg::D;

  v8f acc = {0.f, 0.f, 0.f, 0.f, 0.f, 0.f, 0.f, 0.f};
#pragma unroll
  for (int kb = 0; kb < 4; ++kb) {
    v16h a = Av[kb * 32];
    int d0 = kb * 32 + hi * 8;
    v16h bfrag;
#pragma unroll
    for (int i = 0; i < 8; ++i) {
      bfrag[i]     = (h16)kcol[d0 + i];
      bfrag[8 + i] = (h16)kcol[d0 + 16 + i];
    }
    acc = WMMA_F16(a, bfrag, acc);
  }

  int e0 = mt * 16 + hi * 8;
  size_t base = ((size_t)bn * (cfg::T * cfg::HW) + colq) * cfg::D + e0;
#pragma unroll
  for (int j = 0; j < 8; ++j) Vp[base + j] = acc[j] + bv[n * cfg::D + e0 + j];
}

// ---------------------------------------------------------------------------
// Attention: one wave per (b,n,s,hw). Lane owns dd = 4*lane..4*lane+3.
// Writes y straight into the packed frag layout for the output GEMM.
// ---------------------------------------------------------------------------
__global__ __launch_bounds__(256) void attention_kernel(
    const float* __restrict__ Kp, const float* __restrict__ Qp,
    const float* __restrict__ Vp, h16* __restrict__ Ypk) {
  int lane = threadIdx.x & 31;
  int w    = threadIdx.x >> 5;
  int wg   = blockIdx.x * 8 + w;         // 0..75263
  int hw   = wg % cfg::HW;
  int tmp  = wg / cfg::HW;
  int s    = tmp % cfg::S;
  int bn   = tmp / cfg::S;
  int dd0  = lane * 4;

  size_t pixStride = (size_t)cfg::D;
  size_t rowBase   = ((size_t)bn * cfg::T) * cfg::HW;  // + t*49 + hw, then *D

  v4f qv = *(const v4f*)(Qp + (rowBase + (size_t)(s + 4) * cfg::HW + hw) * pixStride + dd0);

  float al[cfg::KW];
#pragma unroll
  for (int k = 0; k < cfg::KW; ++k) {
    v4f kv = *(const v4f*)(Kp + (rowBase + (size_t)(s + k) * cfg::HW + hw) * pixStride + dd0);
    float p = qv[0] * kv[0] + qv[1] * kv[1] + qv[2] * kv[2] + qv[3] * kv[3];
#pragma unroll
    for (int off = 16; off > 0; off >>= 1) p += __shfl_xor(p, off, 32);
    al[k] = p;
  }
  float mx = al[0];
#pragma unroll
  for (int k = 1; k < cfg::KW; ++k) mx = fmaxf(mx, al[k]);
  float sum = 0.f;
#pragma unroll
  for (int k = 0; k < cfg::KW; ++k) { al[k] = __expf(al[k] - mx); sum += al[k]; }
  float inv = 1.f / sum;

  v4f acc = {0.f, 0.f, 0.f, 0.f};
#pragma unroll
  for (int k = 0; k < cfg::KW; ++k) {
    v4f vv = *(const v4f*)(Vp + (rowBase + (size_t)(s + k) * cfg::HW + hw) * pixStride + dd0);
    float wk = al[k] * inv;
    acc[0] += wk * vv[0]; acc[1] += wk * vv[1];
    acc[2] += wk * vv[2]; acc[3] += wk * vv[3];
  }

  // scatter into B-frag layout for output GEMM: column p2, rows k = n*128+dd
  int b   = bn >> 3;
  int n   = bn & 7;
  int p2  = (b * cfg::S + s) * cfg::HW + hw;
  int nt2 = p2 >> 4;
  int ci  = p2 & 15;
#pragma unroll
  for (int j = 0; j < 4; ++j) {
    int k  = n * cfg::D + dd0 + j;
    int kb = k >> 5;
    int v  = k & 31;
    int hi2 = (v < 16) ? (v >> 3) : ((v >= 24) ? 1 : 0);
    int ii  = (v < 16) ? (v & 7) : ((v & 7) + 8);
    size_t idx = (((size_t)(nt2 * cfg::KB + kb) * 32) + hi2 * 16 + ci) * 16 + ii;
    Ypk[idx] = (h16)acc[j];
  }
}

// ---------------------------------------------------------------------------
// GEMM 3: Z = WoFull @ Y  + bo.sum(0) + 8*x_item   (fused epilogue)
// 2 M-tiles x 4 N-tiles per wave: 6 frag loads -> 8 WMMAs per k-step.
// grid (147, 4), block 256
// ---------------------------------------------------------------------------
__global__ __launch_bounds__(256) void gemm_out_kernel(
    const h16* __restrict__ ApkO, const h16* __restrict__ Ypk,
    const float* __restrict__ boSum, const float* __restrict__ x,
    float* __restrict__ out) {
  int lane = threadIdx.x & 31;
  int w    = threadIdx.x >> 5;
  int mtp  = blockIdx.y * 8 + w;         // 0..31
  int mt0  = mtp * 2, mt1 = mt0 + 1;
  int nta  = blockIdx.x * 4;             // 4 consecutive N tiles

  const v16h* Ao0 = ((const v16h*)ApkO) + (size_t)mt0 * (cfg::KB * 32) + lane;
  const v16h* Ao1 = ((const v16h*)ApkO) + (size_t)mt1 * (cfg::KB * 32) + lane;
  const v16h* Y0  = ((const v16h*)Ypk)  + (size_t)(nta + 0) * (cfg::KB * 32) + lane;
  const v16h* Y1  = ((const v16h*)Ypk)  + (size_t)(nta + 1) * (cfg::KB * 32) + lane;
  const v16h* Y2  = ((const v16h*)Ypk)  + (size_t)(nta + 2) * (cfg::KB * 32) + lane;
  const v16h* Y3  = ((const v16h*)Ypk)  + (size_t)(nta + 3) * (cfg::KB * 32) + lane;

  v8f a00 = {0.f,0.f,0.f,0.f,0.f,0.f,0.f,0.f};
  v8f a01 = a00, a02 = a00, a03 = a00, a10 = a00, a11 = a00, a12 = a00, a13 = a00;

#pragma unroll 2
  for (int kb = 0; kb < cfg::KB; ++kb) {
    v16h b0 = Y0[kb * 32];
    v16h b1 = Y1[kb * 32];
    v16h b2 = Y2[kb * 32];
    v16h b3 = Y3[kb * 32];
    v16h a0 = Ao0[kb * 32];
    v16h a1 = Ao1[kb * 32];
    a00 = WMMA_F16(a0, b0, a00);
    a01 = WMMA_F16(a0, b1, a01);
    a02 = WMMA_F16(a0, b2, a02);
    a03 = WMMA_F16(a0, b3, a03);
    a10 = WMMA_F16(a1, b0, a10);
    a11 = WMMA_F16(a1, b1, a11);
    a12 = WMMA_F16(a1, b2, a12);
    a13 = WMMA_F16(a1, b3, a13);
  }

  store_out_tile(out, x, boSum, a00, mt0, nta + 0, lane);
  store_out_tile(out, x, boSum, a01, mt0, nta + 1, lane);
  store_out_tile(out, x, boSum, a02, mt0, nta + 2, lane);
  store_out_tile(out, x, boSum, a03, mt0, nta + 3, lane);
  store_out_tile(out, x, boSum, a10, mt1, nta + 0, lane);
  store_out_tile(out, x, boSum, a11, mt1, nta + 1, lane);
  store_out_tile(out, x, boSum, a12, mt1, nta + 2, lane);
  store_out_tile(out, x, boSum, a13, mt1, nta + 3, lane);
}

// ---------------------------------------------------------------------------
extern "C" void kernel_launch(void* const* d_in, const int* in_sizes, int n_in,
                              void* d_out, int out_size, void* d_ws, size_t ws_size,
                              hipStream_t stream) {
  (void)in_sizes; (void)n_in; (void)out_size; (void)ws_size;
  const float* x  = (const float*)d_in[0];
  const float* Wq = (const float*)d_in[1];
  const float* bq = (const float*)d_in[2];
  const float* Wk = (const float*)d_in[3];
  const float* bk = (const float*)d_in[4];
  const float* Wv = (const float*)d_in[5];
  const float* bv = (const float*)d_in[6];
  const float* Wo = (const float*)d_in[7];
  const float* bo = (const float*)d_in[8];
  float* out = (float*)d_out;

  char* ws = (char*)d_ws;
  h16*   ApkK  = (h16*)(ws + WsOff::ApkK);
  h16*   ApkQ  = (h16*)(ws + WsOff::ApkQ);
  h16*   ApkV  = (h16*)(ws + WsOff::ApkV);
  h16*   ApkO  = (h16*)(ws + WsOff::ApkO);
  float* boSum = (float*)(ws + WsOff::BoSum);
  h16*   Xpk   = (h16*)(ws + WsOff::Xpk);
  float* Kp    = (float*)(ws + WsOff::Kp);
  float* Qp    = (float*)(ws + WsOff::Qp);
  float* Vp    = (float*)(ws + WsOff::Vp);
  h16*   Ypk   = (h16*)(ws + WsOff::Ypk);

  // 1) pack weights + x into WMMA frag layouts (f32 -> f16)
  pack_wkq_kernel<<<4096, 256, 0, stream>>>(Wq, Wk, ApkQ, ApkK);
  pack_wv_kernel<<<512, 256, 0, stream>>>(Wv, ApkV);
  pack_wo_kernel<<<4096, 256, 0, stream>>>(Wo, ApkO);
  bo_sum_kernel<<<4, 256, 0, stream>>>(bo, boSum);
  pack_x_kernel<<<50176, 256, 0, stream>>>(x, Xpk);

  // 2) K/Q projections (dual 1024x1024x12544 GEMM, 2x2 blocking, shared B)
  gemm_kq_kernel<<<dim3(cfg::NT1 / 2, 4), 256, 0, stream>>>(ApkK, ApkQ, Xpk, bk, bq, Kp, Qp);

  // 3) per-head V projection (8 x 128x128x1568 GEMM)
  gemm_v_kernel<<<dim3(cfg::QT, cfg::B * cfg::NH), 256, 0, stream>>>(ApkV, Kp, bv, Vp);

  // 4) 9-tap softmax attention, wave per query, writes packed Y frags
  attention_kernel<<<cfg::B * cfg::NH * cfg::S * cfg::HW / 8, 256, 0, stream>>>(Kp, Qp, Vp, Ypk);

  // 5) output projection 1024x1024x9408, 2x4 blocking, fused bias+residual
  gemm_out_kernel<<<dim3(cfg::NT2 / 4, 4), 256, 0, stream>>>(ApkO, Ypk, boSum, x, out);
}
